// VectorQuantizer2_89094801589102
// MI455X (gfx1250) — matrix operand order repaired and checked by reading, not compiled, not run on hardware
//
#include <hip/hip_runtime.h>

typedef __attribute__((ext_vector_type(16))) _Float16 v16h;
typedef __attribute__((ext_vector_type(8)))  float    v8f;

#define HTOT  16
#define CCH   32
#define BATCH 128
#define VOCAB 4096
#define NELEM (BATCH*CCH*HTOT*HTOT)   // 1048576

#define CHUNK_CODES 64                       // codes staged per LDS chunk
#define NCHUNK      (VOCAB / CHUNK_CODES)    // 64 chunks
#define TILES_PER_BLOCK 16                   // 8 waves x 2 row tiles

// ---------------------------------------------------------------- init
__global__ void vq_init(const float* __restrict__ f, const float* __restrict__ emb,
                        float* __restrict__ f_res, float* __restrict__ f_hat,
                        _Float16* __restrict__ emb_h, float* __restrict__ esq,
                        float* __restrict__ lacc)
{
    int i = blockIdx.x * blockDim.x + threadIdx.x;
    if (i < NELEM) { f_res[i] = f[i]; f_hat[i] = 0.0f; }
    if (i < VOCAB * CCH) emb_h[i] = (_Float16)emb[i];
    if (i < VOCAB) {
        const float* e = emb + (size_t)i * CCH;
        float s = 0.0f;
        #pragma unroll
        for (int c = 0; c < CCH; ++c) s = fmaf(e[c], e[c], s);
        esq[i] = s;
    }
    if (i == 0) *lacc = 0.0f;
}

// ------------------------------------------------- area pooling (downsample)
__global__ void vq_downsample(const float* __restrict__ f_res,
                              _Float16* __restrict__ res, int p)
{
    int t = blockIdx.x * blockDim.x + threadIdx.x;
    int total = BATCH * p * p * CCH;
    if (t >= total) return;
    int c   = t & (CCH - 1);
    int row = t >> 5;
    int x   = row % p;
    int y   = (row / p) % p;
    int b   = row / (p * p);
    int sy = (y * HTOT) / p, ey = ((y + 1) * HTOT + p - 1) / p;
    int sx = (x * HTOT) / p, ex = ((x + 1) * HTOT + p - 1) / p;
    const float* base = f_res + (((size_t)b * CCH + c) * HTOT) * HTOT;
    float s = 0.0f;
    for (int hh = sy; hh < ey; ++hh)
        for (int ww = sx; ww < ex; ++ww)
            s += base[hh * HTOT + ww];
    s *= 1.0f / (float)((ey - sy) * (ex - sx));
    res[(size_t)row * CCH + c] = (_Float16)s;
}

// ------------------------------------------------- WMMA nearest-code search
// Block = 8 waves; each wave owns 2 row tiles (A in regs, reused across the
// whole codebook). Codebook streamed through LDS in 4KB chunks, software-
// pipelined. Compute path is branch-free (tail waves recompute the clamped
// last tile) so EXEC is all-ones for every WMMA; only writes are guarded.
__global__ void __launch_bounds__(256)
vq_argmin(const _Float16* __restrict__ resA, const _Float16* __restrict__ embB,
          const float* __restrict__ esq, int* __restrict__ idxOut, int nTiles)
{
    __shared__ _Float16 sB[2][CHUNK_CODES * CCH];   // 2 x 4 KB double buffer

    int tid  = threadIdx.x;
    int wave = tid >> 5;
    int lane = tid & 31;
    int sub  = lane & 15;
    int hi   = lane >> 4;

    int t0 = blockIdx.x * TILES_PER_BLOCK + wave * 2;
    int t1 = t0 + 1;
    bool own0 = (t0 < nTiles);                 // used only to guard final writes
    bool own1 = (t1 < nTiles);
    int tc0 = min(t0, nTiles - 1);             // clamped -> always valid A data
    int tc1 = min(t1, nTiles - 1);

    // A tiles (16x32 f16): lanes 0-15 hold K{0..7,16..23}, lanes 16-31 K{8..15,24..31}
    union AV { v16h v; float4 q[2]; };
    AV a0, a1;
    {
        const float4* ap = (const float4*)(resA + (size_t)(tc0 * 16 + sub) * CCH + hi * 8);
        a0.q[0] = ap[0]; a0.q[1] = ap[2];
    }
    {
        const float4* ap = (const float4*)(resA + (size_t)(tc1 * 16 + sub) * CCH + hi * 8);
        a1.q[0] = ap[0]; a1.q[1] = ap[2];
    }

    float best0[8], best1[8];
    int   bidx0[8], bidx1[8];
    #pragma unroll
    for (int r = 0; r < 8; ++r) {
        best0[r] = 3.4e38f; bidx0[r] = 0;
        best1[r] = 3.4e38f; bidx1[r] = 0;
    }

    // stage chunk 0 (256 x float4 = 4KB, one 16B load/store per thread)
    ((float4*)sB[0])[tid] = ((const float4*)embB)[tid];
    __syncthreads();

    for (int ch = 0; ch < NCHUNK; ++ch) {
        int cur = ch & 1;

        // issue next chunk's global load early (latency hidden behind WMMAs)
        float4 stage = float4{};
        if (ch + 1 < NCHUNK)
            stage = ((const float4*)(embB + (size_t)(ch + 1) * CHUNK_CODES * CCH))[tid];

        #pragma unroll
        for (int ct = 0; ct < 4; ++ct) {
            int code = ch * CHUNK_CODES + ct * 16 + sub;   // column N = lane&15
            // B (32x16 f16) from LDS: lanes 0-15 hold K{0..7,16..23}, lanes 16-31 K{8..15,24..31}
            union AV bm;
            const float4* bp = (const float4*)(sB[cur] + (ct * 16 + sub) * CCH + hi * 8);
            bm.q[0] = bp[0];
            bm.q[1] = bp[2];
            float es = esq[code];

            {
                v8f cz = {};
                v8f d = __builtin_amdgcn_wmma_f32_16x16x32_f16(
                            false, a0.v, false, bm.v, (short)0, cz, false, false);
                #pragma unroll
                for (int r = 0; r < 8; ++r) {
                    float s = fmaf(-2.0f, d[r], es);
                    if (s < best0[r]) { best0[r] = s; bidx0[r] = code; }
                }
            }
            {
                v8f cz = {};
                v8f d = __builtin_amdgcn_wmma_f32_16x16x32_f16(
                            false, a1.v, false, bm.v, (short)0, cz, false, false);
                #pragma unroll
                for (int r = 0; r < 8; ++r) {
                    float s = fmaf(-2.0f, d[r], es);
                    if (s < best1[r]) { best1[r] = s; bidx1[r] = code; }
                }
            }
        }

        if (ch + 1 < NCHUNK)
            ((float4*)sB[cur ^ 1])[tid] = stage;
        __syncthreads();
    }

    // min-reduce across the 16 column lanes (xor 1,2,4,8 stays in half-wave)
    #pragma unroll
    for (int off = 1; off < 16; off <<= 1) {
        #pragma unroll
        for (int r = 0; r < 8; ++r) {
            float os; int oi;
            os = __shfl_xor(best0[r], off, 32);
            oi = __shfl_xor(bidx0[r], off, 32);
            if (os < best0[r] || (os == best0[r] && oi < bidx0[r])) { best0[r] = os; bidx0[r] = oi; }
            os = __shfl_xor(best1[r], off, 32);
            oi = __shfl_xor(bidx1[r], off, 32);
            if (os < best1[r] || (os == best1[r] && oi < bidx1[r])) { best1[r] = os; bidx1[r] = oi; }
        }
    }
    if (sub == 0) {
        if (own0) {
            int rb = t0 * 16 + hi * 8;
            #pragma unroll
            for (int r = 0; r < 8; ++r) idxOut[rb + r] = bidx0[r];
        }
        if (own1) {
            int rb = t1 * 16 + hi * 8;
            #pragma unroll
            for (int r = 0; r < 8; ++r) idxOut[rb + r] = bidx1[r];
        }
    }
}

// ------------------------------------------------- bicubic upsample + gather
__device__ inline void cubw(float t, float w[4])
{
    const float A = -0.75f;
    float t1 = 1.0f + t;
    w[0] = A * (t1*t1*t1 - 5.0f*t1*t1 + 8.0f*t1 - 4.0f);
    w[1] = (A + 2.0f)*t*t*t - (A + 3.0f)*t*t + 1.0f;
    float u = 1.0f - t;
    w[2] = (A + 2.0f)*u*u*u - (A + 3.0f)*u*u + 1.0f;
    float v = 2.0f - t;
    w[3] = A * (v*v*v - 5.0f*v*v + 8.0f*v - 4.0f);
}

__global__ void vq_upsample(const int* __restrict__ idx, const float* __restrict__ emb,
                            float* __restrict__ h, int p)
{
    int t = blockIdx.x * blockDim.x + threadIdx.x;
    if (t >= NELEM) return;
    int ww = t & 15;
    int hh = (t >> 4) & 15;
    int c  = (t >> 8) & 31;
    int b  = t >> 13;
    float scale = (float)p * (1.0f / 16.0f);

    float wy[4], wx[4];
    int jy[4], jx[4];
    {
        float xf = ((float)hh + 0.5f) * scale - 0.5f;
        int i0 = (int)floorf(xf);
        cubw(xf - (float)i0, wy);
        #pragma unroll
        for (int k = 0; k < 4; ++k) { int j = i0 - 1 + k; jy[k] = min(max(j, 0), p - 1); }
    }
    {
        float xf = ((float)ww + 0.5f) * scale - 0.5f;
        int i0 = (int)floorf(xf);
        cubw(xf - (float)i0, wx);
        #pragma unroll
        for (int k = 0; k < 4; ++k) { int j = i0 - 1 + k; jx[k] = min(max(j, 0), p - 1); }
    }
    const int* ib = idx + b * p * p;
    float s = 0.0f;
    #pragma unroll
    for (int a = 0; a < 4; ++a) {
        int ry = jy[a] * p;
        #pragma unroll
        for (int q = 0; q < 4; ++q) {
            int code = ib[ry + jx[q]];
            s = fmaf(wy[a] * wx[q], emb[(size_t)code * CCH + c], s);
        }
    }
    h[t] = s;   // layout [b, c, hh, ww]
}

// ------------------------------------------------- conv3x3 Phi + state update
__global__ void __launch_bounds__(256)
vq_conv_update(const float* __restrict__ h, const float* __restrict__ conv_w,
               const float* __restrict__ conv_b, const float* __restrict__ f,
               float* __restrict__ f_hat, float* __restrict__ f_res,
               float* __restrict__ partial, int kIdx)
{
    __shared__ float hs[CCH * HTOT * HTOT];   // 32 KB: one image's activations
    __shared__ float red[256];
    int b = blockIdx.x;
    const float* hb = h + (size_t)b * CCH * HTOT * HTOT;
    for (int i = threadIdx.x; i < CCH * HTOT * HTOT; i += 256) hs[i] = hb[i];
    __syncthreads();

    const float* wk = conv_w + (size_t)kIdx * CCH * CCH * 9;
    const float* bk = conv_b + kIdx * CCH;
    float lsum = 0.0f;

    for (int o = threadIdx.x; o < CCH * HTOT * HTOT; o += 256) {
        int x  = o & 15;
        int y  = (o >> 4) & 15;
        int co = o >> 8;
        float acc = bk[co];
        for (int ci = 0; ci < CCH; ++ci) {
            const float* wp = wk + (co * CCH + ci) * 9;
            const float* hp = hs + ci * 256;
            #pragma unroll
            for (int ky = 0; ky < 3; ++ky) {
                int yy = y + ky - 1;
                if (yy < 0 || yy > 15) continue;
                #pragma unroll
                for (int kx = 0; kx < 3; ++kx) {
                    int xx = x + kx - 1;
                    if (xx < 0 || xx > 15) continue;
                    acc = fmaf(wp[ky * 3 + kx], hp[yy * 16 + xx], acc);
                }
            }
        }
        float out = 0.5f * hs[o] + 0.5f * acc;     // Phi with ratio 0.5
        int g = b * (CCH * HTOT * HTOT) + o;
        float fh = f_hat[g] + out;
        f_hat[g] = fh;
        f_res[g] -= out;
        float df = fh - f[g];
        lsum = fmaf(df, df, lsum);
    }
    red[threadIdx.x] = lsum;
    __syncthreads();
    for (int s = 128; s > 0; s >>= 1) {
        if (threadIdx.x < s) red[threadIdx.x] += red[threadIdx.x + s];
        __syncthreads();
    }
    if (threadIdx.x == 0) partial[b] = red[0];
}

// ------------------------------------------------- deterministic loss reduce
__global__ void vq_loss_reduce(const float* __restrict__ partial, float* __restrict__ lacc)
{
    __shared__ float red[128];
    red[threadIdx.x] = partial[threadIdx.x];
    __syncthreads();
    for (int s = 64; s > 0; s >>= 1) {
        if (threadIdx.x < s) red[threadIdx.x] += red[threadIdx.x + s];
        __syncthreads();
    }
    // per-scale contribution: (BETA+1)*mean / L = 1.25 * sum / (1048576*10)
    if (threadIdx.x == 0) *lacc += red[0] * (1.25f / ((float)NELEM * 10.0f));
}

__global__ void vq_finalize(const float* __restrict__ lacc, float* __restrict__ out)
{
    if (threadIdx.x == 0 && blockIdx.x == 0) out[NELEM] = *lacc;
}

// ---------------------------------------------------------------- launcher
extern "C" void kernel_launch(void* const* d_in, const int* in_sizes, int n_in,
                              void* d_out, int out_size, void* d_ws, size_t ws_size,
                              hipStream_t stream)
{
    (void)in_sizes; (void)n_in; (void)out_size; (void)ws_size;

    const float* f      = (const float*)d_in[0];
    const float* emb    = (const float*)d_in[1];
    const float* conv_w = (const float*)d_in[2];
    const float* conv_b = (const float*)d_in[3];
    float* out = (float*)d_out;              // [0..1048575] f_hat, [1048576] loss

    char* ws = (char*)d_ws;
    float*    f_res  = (float*)ws;     ws += (size_t)NELEM * 4;
    float*    hbuf   = (float*)ws;     ws += (size_t)NELEM * 4;
    float*    esq    = (float*)ws;     ws += (size_t)VOCAB * 4;
    _Float16* emb_h  = (_Float16*)ws;  ws += (size_t)VOCAB * CCH * 2;   // 16B-aligned
    _Float16* res_h  = (_Float16*)ws;  ws += (size_t)BATCH * 256 * CCH * 2;
    int*      idxb   = (int*)ws;       ws += (size_t)BATCH * 256 * 4;
    float*    partial= (float*)ws;     ws += 128 * 4;
    float*    lacc   = (float*)ws;     ws += 4;

    vq_init<<<(NELEM + 255) / 256, 256, 0, stream>>>(f, emb, f_res, out, emb_h, esq, lacc);

    static const int PN[10] = {1, 2, 3, 4, 5, 6, 8, 10, 13, 16};
    static const int KI[10] = {0, 0, 0, 1, 1, 2, 2, 3, 3, 3};   // argmin over ticks (static)

    for (int i = 0; i < 10; ++i) {
        int p = PN[i];
        int N = BATCH * p * p;              // rows (always divisible by 16)
        int total = N * CCH;
        vq_downsample<<<(total + 255) / 256, 256, 0, stream>>>(f_res, res_h, p);

        int tiles = N / 16;
        int blocks = (tiles + TILES_PER_BLOCK - 1) / TILES_PER_BLOCK;
        vq_argmin<<<blocks, 256, 0, stream>>>(res_h, emb_h, esq, idxb, tiles);

        vq_upsample<<<(NELEM + 255) / 256, 256, 0, stream>>>(idxb, emb, hbuf, p);
        vq_conv_update<<<BATCH, 256, 0, stream>>>(hbuf, conv_w, conv_b, f,
                                                  out, f_res, partial, KI[i]);
        vq_loss_reduce<<<1, 128, 0, stream>>>(partial, lacc);
    }
    vq_finalize<<<1, 64, 0, stream>>>(lacc, out);
}
